// SelfAttention_50663434223863
// MI455X (gfx1250) — compile-verified, matrix operand order checked by reading, not run on hardware
//
#include <hip/hip_runtime.h>
#include <hip/hip_bf16.h>

// ---------------------------------------------------------------------------
// Causal self-attention block (LN -> QKV -> flash attn -> out proj + resi)
// MI455X / gfx1250: wave32, WMMA f32_16x16x32_f16.
// Layout strategy: weights pre-transposed (wT[n][k]) and V produced
// pre-transposed (vT[(b,h),d,t]) so every WMMA fragment is two contiguous
// 16-byte loads; GEMM + attention hot loops have no LDS transposes/barriers.
// ---------------------------------------------------------------------------

typedef _Float16 h8   __attribute__((ext_vector_type(8)));
typedef _Float16 v16h __attribute__((ext_vector_type(16)));
typedef float    v8f  __attribute__((ext_vector_type(8)));

static constexpr int BB   = 8;
static constexpr int TT   = 2048;
static constexpr int CC   = 512;   // IN_FEAT
static constexpr int DD   = 512;   // D_FEAT
static constexpr int HH   = 8;
static constexpr int DH   = 64;
static constexpr int NTOK = BB * TT;            // 16384
static constexpr float SCALE_INV = 0.125f;      // 1/sqrt(64)
static constexpr float NEGBIG    = -1e9f;

// --- WMMA fragment helpers -------------------------------------------------
// 16-bit A/B fragment: per lane, 8 contiguous f16 at +0 and 8 at +16 elements,
// with element base (0 or 8) selected by lane half.
__device__ __forceinline__ v16h ldfrag(const _Float16* p) {
  h8 lo = *(const h8*)(p);
  h8 hi = *(const h8*)(p + 16);
  return __builtin_shufflevector(lo, hi, 0,1,2,3,4,5,6,7,8,9,10,11,12,13,14,15);
}

__device__ __forceinline__ v8f wmma_f16(v16h a, v16h b, v8f c) {
  return __builtin_amdgcn_wmma_f32_16x16x32_f16(
      false, a, false, b, (short)0, c, false, false);
}

__device__ __forceinline__ float redmax16(float v) {
#pragma unroll
  for (int off = 1; off < 16; off <<= 1) v = fmaxf(v, __shfl_xor(v, off, 32));
  return v;
}
__device__ __forceinline__ float redsum16(float v) {
#pragma unroll
  for (int off = 1; off < 16; off <<= 1) v += __shfl_xor(v, off, 32);
  return v;
}

// ---------------------------------------------------------------------------
// Kernel 1: convert + TRANSPOSE 4 weight matrices (512x512): wT16[n][k]=w[k][n]
// ---------------------------------------------------------------------------
__global__ void wconv_k(const float* __restrict__ s0, const float* __restrict__ s1,
                        const float* __restrict__ s2, const float* __restrict__ s3,
                        _Float16* __restrict__ dst) {
  __shared__ __align__(16) _Float16 tl[64 * 72];   // [n][k] tile, stride 72

  const float* srcs[4] = {s0, s1, s2, s3};
  const float* s = srcs[blockIdx.z];
  _Float16* o = dst + (size_t)blockIdx.z * (CC * DD);

  const int k0 = blockIdx.x * 64;
  const int n0 = blockIdx.y * 64;
  const int tid = threadIdx.x;   // 256

  for (int i = tid; i < 64 * 64; i += 256) {
    int k = i >> 6, n = i & 63;
    tl[n * 72 + k] = (_Float16)s[(size_t)(k0 + k) * DD + n0 + n];
  }
  __syncthreads();
  for (int i = tid; i < 64 * 8; i += 256) {     // 8 x h8 chunks per row
    int row = i >> 3, ch = (i & 7) * 8;
    *(h8*)&o[(size_t)(n0 + row) * CC + k0 + ch] = *(const h8*)&tl[row * 72 + ch];
  }
}

// ---------------------------------------------------------------------------
// Kernel 2: LayerNorm over channels + transpose (B,C,T) -> token-major f16 h
// ---------------------------------------------------------------------------
__global__ void ln_k(const float* __restrict__ x, const float* __restrict__ gamma,
                     const float* __restrict__ beta, _Float16* __restrict__ h16) {
  __shared__ float tile[CC][17];
  __shared__ float mu_s[16], rs_s[16];

  const int tid = threadIdx.x;                 // 256
  const int bt0 = blockIdx.x * 16;
  const int b   = bt0 >> 11;
  const int t0  = bt0 & (TT - 1);

  for (int i = tid; i < CC * 16; i += 256) {
    int c = i >> 4, tk = i & 15;
    tile[c][tk] = x[((size_t)(b * CC + c) << 11) + (t0 + tk)];
  }
  __syncthreads();

  {
    int tok = tid >> 4, seg = tid & 15;
    float sm = 0.f, ss = 0.f;
    for (int c = seg; c < CC; c += 16) {
      float v = tile[c][tok];
      sm += v; ss += v * v;
    }
    sm = redsum16(sm); ss = redsum16(ss);
    if (seg == 0) {
      float mu  = sm * (1.0f / CC);
      float var = ss * (1.0f / CC) - mu * mu;
      mu_s[tok] = mu;
      rs_s[tok] = rsqrtf(var + 1e-5f);
    }
  }
  __syncthreads();

  for (int i = tid; i < CC * 16; i += 256) {
    int tk = i >> 9, c = i & (CC - 1);
    float hv = (tile[c][tk] - mu_s[tk]) * rs_s[tk] * gamma[c] + beta[c];
    h16[((size_t)(bt0 + tk) << 9) + c] = (_Float16)hv;
  }
}

// ---------------------------------------------------------------------------
// Kernel 3: Q/K/V projection GEMM (16384x512)x(512x512)+bias.
// 4 waves x (32M x 64N); A and B fragments straight from global (B = wT).
// z==2 (V) writes transposed layout vT[(b*H+h)*64+d][T] via LDS staging.
// ---------------------------------------------------------------------------
__global__ void qkv_k(const _Float16* __restrict__ h16, const _Float16* __restrict__ wT_all,
                      const float* __restrict__ bq, const float* __restrict__ bk,
                      const float* __restrict__ bv,
                      _Float16* __restrict__ q16, _Float16* __restrict__ k16,
                      _Float16* __restrict__ vT) {
  __shared__ __align__(16) _Float16 smem[128 * 72];  // also viewed as [64][136]

  const int tid  = threadIdx.x;       // 128
  const int w    = tid >> 5;
  const int lane = tid & 31;
  const int nlo  = lane & 15;
  const int basek = (lane < 16) ? 0 : 8;
  const int half8 = (lane < 16) ? 0 : 8;

  const int m0  = blockIdx.x * 128;
  const int n0  = blockIdx.y * 64;
  const int z   = blockIdx.z;
  const int m0w = m0 + w * 32;

  const _Float16* wT = wT_all + (size_t)z * (CC * DD);
  const float* bias = (z == 0) ? bq : (z == 1) ? bk : bv;

  v8f acc[2][4] = {};

  for (int kk = 0; kk < CC; kk += 32) {
    v16h af0 = ldfrag(h16 + ((size_t)(m0w + nlo) << 9) + kk + basek);
    v16h af1 = ldfrag(h16 + ((size_t)(m0w + 16 + nlo) << 9) + kk + basek);
#pragma unroll
    for (int nt = 0; nt < 4; ++nt) {
      v16h bf = ldfrag(wT + (size_t)(n0 + nt * 16 + nlo) * CC + kk + basek);
      acc[0][nt] = wmma_f16(af0, bf, acc[0][nt]);
      acc[1][nt] = wmma_f16(af1, bf, acc[1][nt]);
    }
  }

  if (z < 2) {
    // stage [m_local][n_local] then coalesced b128 writeout (token-major)
#pragma unroll
    for (int nt = 0; nt < 4; ++nt) {
      float bsv = bias[n0 + nt * 16 + nlo];
#pragma unroll
      for (int mh = 0; mh < 2; ++mh)
#pragma unroll
        for (int r = 0; r < 8; ++r)
          smem[(w * 32 + mh * 16 + r + half8) * 72 + nt * 16 + nlo] =
              (_Float16)(acc[mh][nt][r] + bsv);
    }
    __syncthreads();
    _Float16* outp = (z == 0) ? q16 : k16;
    for (int i = tid; i < 128 * 8; i += 128) {
      int row = i >> 3, ch = (i & 7) * 8;
      *(h8*)(outp + ((size_t)(m0 + row) << 9) + n0 + ch) =
          *(const h8*)&smem[row * 72 + ch];
    }
  } else {
    // stage transposed [d_local][m_local]; write vT rows (t-contiguous)
#pragma unroll
    for (int nt = 0; nt < 4; ++nt) {
      float bsv = bias[n0 + nt * 16 + nlo];
#pragma unroll
      for (int mh = 0; mh < 2; ++mh)
#pragma unroll
        for (int r = 0; r < 8; ++r)
          smem[(nt * 16 + nlo) * 136 + (w * 32 + mh * 16 + r + half8)] =
              (_Float16)(acc[mh][nt][r] + bsv);
    }
    __syncthreads();
    const int b = m0 >> 11, t0 = m0 & (TT - 1);
    const int head = n0 >> 6;
    for (int i = tid; i < 64 * 16; i += 128) {
      int row = i >> 4, ch = (i & 15) * 8;
      *(h8*)(vT + ((size_t)((b * HH + head) * DH + row) << 11) + t0 + ch) =
          *(const h8*)&smem[row * 136 + ch];
    }
  }
}

// ---------------------------------------------------------------------------
// Kernel 4: causal flash attention. 4 waves x 16 q-rows; 32-key blocks.
// No block barriers: K fragments from global (token-major), V fragments from
// pre-transposed vT (global), P staged per-wave in LDS (same-wave ordering).
// Each wave iterates only its own causal bound; mask applied on last block.
// ---------------------------------------------------------------------------
__global__ void attn_k(const _Float16* __restrict__ q16, const _Float16* __restrict__ k16,
                       const _Float16* __restrict__ vT, _Float16* __restrict__ ctx16) {
  __shared__ __align__(16) _Float16 Pl[4 * 16 * 40];   // per-wave P tile
  __shared__ __align__(16) _Float16 Ct[4 * 16 * 72];   // per-wave ctx staging

  const int tid  = threadIdx.x;       // 128
  const int w    = tid >> 5;
  const int lane = tid & 31;
  const int nlo  = lane & 15;
  const int basek = (lane < 16) ? 0 : 8;
  const int half8 = (lane < 16) ? 0 : 8;

  const int qi   = blockIdx.x;
  const int head = blockIdx.y;
  const int bz   = blockIdx.z;
  const int bt0  = bz * TT;
  const int qr   = qi * 64 + w * 16;         // wave's first q row
  const int hoff = head * DH;

  const _Float16* vbase = vT + ((size_t)((bz * HH + head) * DH) << 11);

  const _Float16* qp = q16 + ((size_t)(bt0 + qr + nlo) << 9) + hoff + basek;
  v16h aq0 = ldfrag(qp);
  v16h aq1 = ldfrag(qp + 32);

  v8f o[4] = {};
  float mrow[8], lrow[8];
#pragma unroll
  for (int r = 0; r < 8; ++r) { mrow[r] = -__builtin_inff(); lrow[r] = 0.f; }

  const int nbw = (qr >> 5) + 1;             // wave-specific causal bound
  for (int kb = 0; kb < nbw; ++kb) {
    const int k0 = kb * 32;
    const bool maskblk = (kb == nbw - 1);

    if (kb + 1 < nbw) {  // prefetch next key block (K rows + vT columns)
      __builtin_prefetch(k16 + ((size_t)(bt0 + k0 + 32 + nlo) << 9) + hoff, 0, 1);
      __builtin_prefetch(vbase + (((size_t)(2 * lane)) << 11) + k0 + 32, 0, 1);
    }

    // S = Q K^T (two 16-key tiles)
    v8f s0 = {}, s1 = {};
    {
      const _Float16* kp0 = k16 + ((size_t)(bt0 + k0 + nlo) << 9) + hoff + basek;
      s0 = wmma_f16(aq0, ldfrag(kp0), s0);
      s0 = wmma_f16(aq1, ldfrag(kp0 + 32), s0);
      const _Float16* kp1 = k16 + ((size_t)(bt0 + k0 + 16 + nlo) << 9) + hoff + basek;
      s1 = wmma_f16(aq0, ldfrag(kp1), s1);
      s1 = wmma_f16(aq1, ldfrag(kp1 + 32), s1);
    }

    // online softmax
    const int key0 = k0 + nlo;
    const int key1 = k0 + 16 + nlo;
#pragma unroll
    for (int r = 0; r < 8; ++r) {
      float sv0 = s0[r] * SCALE_INV;
      float sv1 = s1[r] * SCALE_INV;
      if (maskblk) {
        int qrow = qr + r + half8;
        if (key0 > qrow) sv0 += NEGBIG;
        if (key1 > qrow) sv1 += NEGBIG;
      }
      float mt = redmax16(fmaxf(sv0, sv1));
      float mold = mrow[r];
      float mn = fmaxf(mold, mt);
      float alpha = __expf(mold - mn);
      float p0 = __expf(sv0 - mn);
      float p1 = __expf(sv1 - mn);
      float rs = redsum16(p0 + p1);
      lrow[r] = lrow[r] * alpha + rs;
      mrow[r] = mn;
#pragma unroll
      for (int nt = 0; nt < 4; ++nt) o[nt][r] *= alpha;
      Pl[w * 640 + (r + half8) * 40 + nlo]      = (_Float16)p0;
      Pl[w * 640 + (r + half8) * 40 + 16 + nlo] = (_Float16)p1;
    }

    // O += P * V : P fragment from per-wave LDS, V fragment from global vT
    v16h pf = ldfrag(&Pl[w * 640 + nlo * 40 + basek]);
#pragma unroll
    for (int nt = 0; nt < 4; ++nt) {
      v16h bv = ldfrag(vbase + ((size_t)(nt * 16 + nlo) << 11) + k0 + basek);
      o[nt] = wmma_f16(pf, bv, o[nt]);
    }
  }

  // finalize: O /= l, stage per-wave then b128 writeout (token-major ctx)
#pragma unroll
  for (int r = 0; r < 8; ++r) {
    float inv = 1.0f / lrow[r];
#pragma unroll
    for (int nt = 0; nt < 4; ++nt)
      Ct[w * 1152 + (r + half8) * 72 + nt * 16 + nlo] = (_Float16)(o[nt][r] * inv);
  }
  for (int i = lane; i < 16 * 8; i += 32) {
    int row = i >> 3, ch = (i & 7) * 8;
    *(h8*)(ctx16 + ((size_t)(bt0 + qr + row) << 9) + hoff + ch) =
        *(const h8*)&Ct[w * 1152 + row * 72 + ch];
  }
}

// ---------------------------------------------------------------------------
// Kernel 5: out = resi + (ctx @ wo + bo), transposed to (B, C, T).
// Same fragment-from-global GEMM; f32 LDS staging for coalesced t-writes.
// ---------------------------------------------------------------------------
__global__ void outproj_k(const _Float16* __restrict__ ctx16, const _Float16* __restrict__ woT,
                          const float* __restrict__ bo, const float* __restrict__ x,
                          float* __restrict__ out) {
  __shared__ float cf[64 * 129];   // [c_local][tok_local]

  const int tid  = threadIdx.x;    // 128
  const int w    = tid >> 5;
  const int lane = tid & 31;
  const int nlo  = lane & 15;
  const int basek = (lane < 16) ? 0 : 8;
  const int half8 = (lane < 16) ? 0 : 8;

  const int m0  = blockIdx.x * 128;
  const int n0  = blockIdx.y * 64;
  const int m0w = m0 + w * 32;

  v8f acc[2][4] = {};

  for (int kk = 0; kk < DD; kk += 32) {
    v16h af0 = ldfrag(ctx16 + ((size_t)(m0w + nlo) << 9) + kk + basek);
    v16h af1 = ldfrag(ctx16 + ((size_t)(m0w + 16 + nlo) << 9) + kk + basek);
#pragma unroll
    for (int nt = 0; nt < 4; ++nt) {
      v16h bf = ldfrag(woT + (size_t)(n0 + nt * 16 + nlo) * DD + kk + basek);
      acc[0][nt] = wmma_f16(af0, bf, acc[0][nt]);
      acc[1][nt] = wmma_f16(af1, bf, acc[1][nt]);
    }
  }

#pragma unroll
  for (int nt = 0; nt < 4; ++nt) {
    int c_local = nt * 16 + nlo;
    float bsv = bo[n0 + c_local];
#pragma unroll
    for (int mh = 0; mh < 2; ++mh)
#pragma unroll
      for (int r = 0; r < 8; ++r)
        cf[c_local * 129 + (w * 32 + mh * 16 + r + half8)] = acc[mh][nt][r] + bsv;
  }
  __syncthreads();

  const int b = m0 >> 11, t0 = m0 & (TT - 1);
  for (int i = tid; i < 64 * 128; i += 128) {
    int c_local = i >> 7, tok = i & 127;
    size_t oi = ((size_t)(b * CC + n0 + c_local) << 11) + t0 + tok;
    out[oi] = x[oi] + cf[c_local * 129 + tok];
  }
}

// ---------------------------------------------------------------------------
extern "C" void kernel_launch(void* const* d_in, const int* in_sizes, int n_in,
                              void* d_out, int out_size, void* d_ws, size_t ws_size,
                              hipStream_t stream) {
  (void)in_sizes; (void)n_in; (void)out_size; (void)ws_size;

  const float* x     = (const float*)d_in[0];
  const float* gamma = (const float*)d_in[1];
  const float* beta  = (const float*)d_in[2];
  const float* wq    = (const float*)d_in[3];
  const float* bq    = (const float*)d_in[4];
  const float* wk    = (const float*)d_in[5];
  const float* bk    = (const float*)d_in[6];
  const float* wv    = (const float*)d_in[7];
  const float* bv    = (const float*)d_in[8];
  const float* wo    = (const float*)d_in[9];
  const float* bo    = (const float*)d_in[10];
  float* out = (float*)d_out;

  // workspace (f16): h | q | k | vT | ctx | wT[q,k,v,o]
  _Float16* h16   = (_Float16*)d_ws;
  _Float16* q16   = h16  + (size_t)NTOK * CC;
  _Float16* k16   = q16  + (size_t)NTOK * DD;
  _Float16* vT    = k16  + (size_t)NTOK * DD;
  _Float16* ctx16 = vT   + (size_t)NTOK * DD;
  _Float16* wT    = ctx16 + (size_t)NTOK * DD;
  _Float16* woT   = wT + (size_t)3 * CC * DD;

  wconv_k  <<<dim3(8, 8, 4), 256, 0, stream>>>(wq, wk, wv, wo, wT);
  ln_k     <<<dim3(NTOK / 16, 1, 1), 256, 0, stream>>>(x, gamma, beta, h16);
  qkv_k    <<<dim3(NTOK / 128, DD / 64, 3), 128, 0, stream>>>(h16, wT, bq, bk, bv, q16, k16, vT);
  attn_k   <<<dim3(TT / 64, HH, BB), 128, 0, stream>>>(q16, k16, vT, ctx16);
  outproj_k<<<dim3(NTOK / 128, CC / 64, 1), 128, 0, stream>>>(ctx16, woT, bo, x, out);
}
